// RGCNModel_48911087567468
// MI455X (gfx1250) — compile-verified, matrix operand order checked by reading, not run on hardware
//
#include <hip/hip_runtime.h>
#include <hip/hip_bf16.h>

typedef __attribute__((ext_vector_type(16))) _Float16 v16h;
typedef __attribute__((ext_vector_type(8)))  _Float16 v8h;
typedef __attribute__((ext_vector_type(8)))  float    v8f;

#define R_REL 8

// ---------------------------------------------------------------------------
// Utility: fp32 -> fp16 conversion (vectorized)
// ---------------------------------------------------------------------------
__global__ __launch_bounds__(256) void f32_to_f16_kernel(
    const float* __restrict__ in, _Float16* __restrict__ out, int n) {
  int i = (blockIdx.x * blockDim.x + threadIdx.x) * 4;
  if (i + 3 < n) {
    float4 v = *(const float4*)(in + i);
    out[i + 0] = (_Float16)v.x;
    out[i + 1] = (_Float16)v.y;
    out[i + 2] = (_Float16)v.z;
    out[i + 3] = (_Float16)v.w;
  } else {
    for (; i < n; ++i) out[i] = (_Float16)in[i];
  }
}

// ---------------------------------------------------------------------------
// Build transposed, concatenated f16 weight matrix BT[9F x K]:
//   row n: block t = n / F   (t==0: W_root, t>=1: W[t-1]), col c = n % F
//   BT[n*K + k] = (t==0 ? root[k*F+c] : W[((t-1)*K + k)*F + c])
// ---------------------------------------------------------------------------
__global__ __launch_bounds__(256) void prep_wt_kernel(
    const float* __restrict__ W, const float* __restrict__ root,
    _Float16* __restrict__ BT, int K, int F, int total) {
  int idx = blockIdx.x * blockDim.x + threadIdx.x;
  if (idx >= total) return;
  int n = idx / K, k = idx - n * K;
  int t = n / F, c = n - t * F;
  float v = (t == 0) ? root[(size_t)k * F + c]
                     : W[((size_t)(t - 1) * K + k) * F + c];
  BT[idx] = (_Float16)v;
}

// ---------------------------------------------------------------------------
// q = elu(question_embedding @ qn_W + qn_b), stored as f16 [B x 32]
// ---------------------------------------------------------------------------
__global__ __launch_bounds__(256) void q_kernel(
    const float* __restrict__ qe, const float* __restrict__ Wq,
    const float* __restrict__ bq, _Float16* __restrict__ q16,
    int B, int QDIM, int RQ) {
  int idx = blockIdx.x * blockDim.x + threadIdx.x;
  if (idx >= B * RQ) return;
  int b = idx / RQ, c = idx - b * RQ;
  float s = bq[c];
  for (int k = 0; k < QDIM; ++k) s += qe[(size_t)b * QDIM + k] * Wq[(size_t)k * RQ + c];
  s = (s > 0.0f) ? s : (__expf(s) - 1.0f);
  q16[idx] = (_Float16)s;
}

// ---------------------------------------------------------------------------
// Per-(dst, relation) edge counts (graph-only; computed once, reused 5x)
// ---------------------------------------------------------------------------
__global__ __launch_bounds__(256) void count_kernel(
    const int* __restrict__ dst, const int* __restrict__ et,
    int* __restrict__ cnt, int E) {
  int e = blockIdx.x * blockDim.x + threadIdx.x;
  if (e < E) atomicAdd(&cnt[(size_t)dst[e] * R_REL + et[e]], 1);
}

__global__ __launch_bounds__(256) void inv_kernel(
    const int* __restrict__ cnt, float* __restrict__ inv, int n) {
  int i = blockIdx.x * blockDim.x + threadIdx.x;
  if (i < n) {
    int c = cnt[i];
    inv[i] = 1.0f / (float)(c > 0 ? c : 1);
  }
}

// ---------------------------------------------------------------------------
// WMMA GEMM:  Z[M x Ncols] = A[M x K] (f16) @ BT[Ncols x K]^T (f16), fp32 acc.
// One wave -> 16 rows x 4 col-tiles (A fragment reused 4x). 8 waves/block.
// The K-loop is fully branchless: out-of-range column tiles are CLAMPED to
// the last valid 16-row tile of BT (Ncols is a multiple of 16) and their
// results are discarded at store time, so accumulators never cross a branch.
// ---------------------------------------------------------------------------
__global__ __launch_bounds__(256) void gemm_wmma_kernel(
    const _Float16* __restrict__ A, const _Float16* __restrict__ BT,
    float* __restrict__ Z, int M, int K, int Ncols) {
  const int lane = threadIdx.x & 31;
  const int wave = threadIdx.x >> 5;
  const int m0 = (blockIdx.y * 8 + wave) * 16;
  if (m0 >= M) return;                         // wave-uniform
  const int nbase = blockIdx.x * 64;
  const int l15 = lane & 15;
  const int koff = (lane >> 4) * 8;            // 0 or 8

  int arow = m0 + l15;
  if (arow >= M) arow = M - 1;                 // safety (M is multiple of 16)

  const _Float16* pa = A + (size_t)arow * K + koff;
  const _Float16* pb[4];
#pragma unroll
  for (int j = 0; j < 4; ++j) {
    int rr = nbase + j * 16;
    if (rr + 16 > Ncols) rr = Ncols - 16;      // clamp to last valid tile
    pb[j] = BT + (size_t)(rr + l15) * K + koff;
  }

  v8f acc[4] = {};

  for (int k0 = 0; k0 < K; k0 += 32) {
    v8h alo = *(const v8h*)(pa + k0);
    v8h ahi = *(const v8h*)(pa + k0 + 16);
    v16h af = __builtin_shufflevector(alo, ahi, 0, 1, 2, 3, 4, 5, 6, 7,
                                      8, 9, 10, 11, 12, 13, 14, 15);
#pragma unroll
    for (int j = 0; j < 4; ++j) {
      v8h blo = *(const v8h*)(pb[j] + k0);
      v8h bhi = *(const v8h*)(pb[j] + k0 + 16);
      v16h bf = __builtin_shufflevector(blo, bhi, 0, 1, 2, 3, 4, 5, 6, 7,
                                        8, 9, 10, 11, 12, 13, 14, 15);
      acc[j] = __builtin_amdgcn_wmma_f32_16x16x32_f16(
          false, af, false, bf, (short)0, acc[j], false, false);
    }
  }

  // C/D layout: VGPR r -> row m0 + r + 8*(lane>=16), col n0 + (lane&15)
  const int rbase = m0 + koff;                 // koff == 8*(lane>=16)
#pragma unroll
  for (int j = 0; j < 4; ++j) {
    int n0 = nbase + j * 16;
    if (n0 < Ncols) {                          // block-uniform
      int col = n0 + l15;
#pragma unroll
      for (int r = 0; r < 8; ++r) {
        int row = rbase + r;
        if (row < M) Z[(size_t)row * Ncols + col] = acc[j][r];
      }
    }
  }
}

// ---------------------------------------------------------------------------
// Edge scatter: acc[dst] += Z[src, (1+r)*F : (2+r)*F] * inv[dst, r]
// One wave per edge; lane l handles float4 at offset 4l.
// ---------------------------------------------------------------------------
__global__ __launch_bounds__(256) void edge_agg_kernel(
    const int* __restrict__ src, const int* __restrict__ dst,
    const int* __restrict__ et, const float* __restrict__ Z,
    const float* __restrict__ inv, float* __restrict__ acc,
    int E, int F, int ldz) {
  int e = blockIdx.x * 8 + (threadIdx.x >> 5);
  if (e >= E) return;                          // wave-uniform
  int lane = threadIdx.x & 31;
  int s = src[e], d = dst[e], r = et[e];
  float w = inv[(size_t)d * R_REL + r];
  const float* zp = Z + (size_t)s * ldz + (size_t)(r + 1) * F;
  float* ap = acc + (size_t)d * F;
  int i = lane * 4;
  if (i < F) {
    float4 v = *(const float4*)(zp + i);
    atomicAdd(ap + i + 0, v.x * w);
    atomicAdd(ap + i + 1, v.y * w);
    atomicAdd(ap + i + 2, v.z * w);
    atomicAdd(ap + i + 3, v.w * w);
  }
}

// ---------------------------------------------------------------------------
// Finalize: v = Z_root[n,j] + bias[j] + acc[n,j]; optional ELU;
// write f16 next-layer input (ld = ldnext) and/or fp32 output.
// ---------------------------------------------------------------------------
__global__ __launch_bounds__(256) void finalize_kernel(
    const float* __restrict__ Z, const float* __restrict__ acc,
    const float* __restrict__ bias, _Float16* __restrict__ Anext,
    float* __restrict__ outf, int N, int F, int ldz, int ldnext, int do_elu) {
  int idx = blockIdx.x * blockDim.x + threadIdx.x;
  if (idx >= N * F) return;
  int n = idx / F, j = idx - n * F;
  float v = Z[(size_t)n * ldz + j] + bias[j] + acc[(size_t)n * F + j];
  if (do_elu) v = (v > 0.0f) ? v : (__expf(v) - 1.0f);
  if (Anext) Anext[(size_t)n * ldnext + j] = (_Float16)v;
  if (outf) outf[(size_t)n * F + j] = v;
}

// Concat q[batch[n]] into cols 96..127 of the layer-1 input (f16, ld=128)
__global__ __launch_bounds__(256) void concat_q_kernel(
    const _Float16* __restrict__ q16, const int* __restrict__ batch,
    _Float16* __restrict__ Anext, int N) {
  int idx = blockIdx.x * blockDim.x + threadIdx.x;
  if (idx >= N * 32) return;
  int n = idx >> 5, c = idx & 31;
  Anext[(size_t)n * 128 + 96 + c] = q16[(size_t)batch[n] * 32 + c];
}

// ---------------------------------------------------------------------------
extern "C" void kernel_launch(void* const* d_in, const int* in_sizes, int n_in,
                              void* d_out, int out_size, void* d_ws, size_t ws_size,
                              hipStream_t stream) {
  (void)n_in; (void)out_size; (void)ws_size;
  // --- inputs (setup_inputs order) ---
  const float* x        = (const float*)d_in[0];
  const float* qe       = (const float*)d_in[1];
  const int*   eidx     = (const int*)d_in[2];
  const int*   eattr    = (const int*)d_in[3];
  const int*   batch    = (const int*)d_in[4];
  const float* qn_W     = (const float*)d_in[5];
  const float* qn_b     = (const float*)d_in[6];
  const float* c0_W     = (const float*)d_in[7];
  const float* c0_root  = (const float*)d_in[8];
  const float* c0_b     = (const float*)d_in[9];
  const float* c1_W     = (const float*)d_in[10];
  const float* c1_root  = (const float*)d_in[11];
  const float* c1_b     = (const float*)d_in[12];
  const float* cm_W     = (const float*)d_in[13];
  const float* cm_root  = (const float*)d_in[14];
  const float* cm_b     = (const float*)d_in[15];
  const float* co_W     = (const float*)d_in[16];
  const float* co_root  = (const float*)d_in[17];
  const float* co_b     = (const float*)d_in[18];

  const int QDIM = 768, RQ = 32, RN = 96, H = 128, NODE = 512;
  const int N = in_sizes[4];
  const int E = in_sizes[3];
  const int B = in_sizes[1] / QDIM;
  const int* esrc = eidx;
  const int* edst = eidx + E;

  // --- workspace layout ---
  auto al = [](size_t v) { return (v + 255) & ~(size_t)255; };
  char* ws = (char*)d_ws;
  size_t off = 0;
  _Float16* A16 = (_Float16*)(ws + off); off = al(off + (size_t)N * NODE * 2);
  float*    Z   = (float*)(ws + off);    off = al(off + (size_t)N * 9 * H * 4);
  float*    acc = (float*)(ws + off);    off = al(off + (size_t)N * H * 4);
  int*      cnt = (int*)(ws + off);      off = al(off + (size_t)N * R_REL * 4);
  float*    inv = (float*)(ws + off);    off = al(off + (size_t)N * R_REL * 4);
  _Float16* BT0 = (_Float16*)(ws + off); off = al(off + (size_t)9 * RN * NODE * 2);
  _Float16* BT1 = (_Float16*)(ws + off); off = al(off + (size_t)9 * H * H * 2);
  _Float16* BT2 = (_Float16*)(ws + off); off = al(off + (size_t)9 * H * H * 2);
  _Float16* BT3 = (_Float16*)(ws + off); off = al(off + (size_t)9 * H * H * 2);
  _Float16* BT4 = (_Float16*)(ws + off); off = al(off + (size_t)9 * H * H * 2);
  _Float16* q16 = (_Float16*)(ws + off); off = al(off + (size_t)B * RQ * 2);

  // --- one-time prep: weights (f16, transposed), q, x->f16, edge counts ---
  {
    int t0 = 9 * RN * NODE, t1 = 9 * H * H;
    prep_wt_kernel<<<(t0 + 255) / 256, 256, 0, stream>>>(c0_W, c0_root, BT0, NODE, RN, t0);
    prep_wt_kernel<<<(t1 + 255) / 256, 256, 0, stream>>>(c1_W, c1_root, BT1, H, H, t1);
    prep_wt_kernel<<<(t1 + 255) / 256, 256, 0, stream>>>(cm_W, cm_root, BT2, H, H, t1);
    prep_wt_kernel<<<(t1 + 255) / 256, 256, 0, stream>>>(cm_W + (size_t)R_REL * H * H,
                                                         cm_root + (size_t)H * H, BT3, H, H, t1);
    prep_wt_kernel<<<(t1 + 255) / 256, 256, 0, stream>>>(co_W, co_root, BT4, H, H, t1);
    q_kernel<<<(B * RQ + 255) / 256, 256, 0, stream>>>(qe, qn_W, qn_b, q16, B, QDIM, RQ);
    int nx = N * NODE;
    f32_to_f16_kernel<<<((nx + 3) / 4 + 255) / 256, 256, 0, stream>>>(x, A16, nx);
    hipMemsetAsync(cnt, 0, (size_t)N * R_REL * 4, stream);
    count_kernel<<<(E + 255) / 256, 256, 0, stream>>>(edst, eattr, cnt, E);
    inv_kernel<<<(N * R_REL + 255) / 256, 256, 0, stream>>>(cnt, inv, N * R_REL);
  }

  const int mblocks = ((N + 15) / 16 + 7) / 8;  // 128 rows per block

  struct Layer { const _Float16* BT; const float* bias; int K, F; int last; };
  Layer layers[5] = {
      {BT0, c0_b, NODE, RN, 0},
      {BT1, c1_b, H, H, 0},
      {BT2, cm_b, H, H, 0},
      {BT3, cm_b + H, H, H, 0},
      {BT4, co_b, H, H, 1},
  };

  for (int L = 0; L < 5; ++L) {
    const Layer& ly = layers[L];
    const int ncols = 9 * ly.F;
    const int ldz = ncols;

    // Z = A16 @ [root | W_0..W_7]   (f16 WMMA, fp32 accumulate)
    dim3 ggrid((ncols + 63) / 64, mblocks);
    gemm_wmma_kernel<<<ggrid, 256, 0, stream>>>(A16, ly.BT, Z, N, ly.K, ncols);

    // per-relation mean aggregation, folded to per-edge scale
    hipMemsetAsync(acc, 0, (size_t)N * ly.F * 4, stream);
    edge_agg_kernel<<<(E + 7) / 8, 256, 0, stream>>>(esrc, edst, eattr, Z, inv,
                                                     acc, E, ly.F, ldz);

    // finalize: root + bias + agg, ELU (except last), emit next input / output
    if (ly.last) {
      finalize_kernel<<<((size_t)N * ly.F + 255) / 256, 256, 0, stream>>>(
          Z, acc, ly.bias, nullptr, (float*)d_out, N, ly.F, ldz, 0, 0);
    } else {
      finalize_kernel<<<((size_t)N * ly.F + 255) / 256, 256, 0, stream>>>(
          Z, acc, ly.bias, A16, nullptr, N, ly.F, ldz, H, 1);
      if (L == 0)
        concat_q_kernel<<<(N * 32 + 255) / 256, 256, 0, stream>>>(q16, batch, A16, N);
    }
  }
}
